// MultiHeadAttention_22574348108504
// MI455X (gfx1250) — matrix operand order repaired and checked by reading, not compiled, not run on hardware
//
#include <hip/hip_runtime.h>
#include <math.h>

#define DIMC  768
#define NHEAD 12
#define NB    8
#define NTOK  1024
#define DHEAD 64
#define QKVC  (3 * DIMC)   // 2304

typedef __attribute__((ext_vector_type(16))) __bf16       v16bf;
typedef __attribute__((ext_vector_type(8)))  float        v8f;
typedef __attribute__((ext_vector_type(4)))  unsigned int u32x4;
typedef __attribute__((ext_vector_type(4)))  float        f32x4;

union AFrag { v16bf v; u32x4 q[2]; };
union Acc8  { v8f v; f32x4 f[2]; };

static __device__ __forceinline__ v8f wmma_bf16(v16bf a, v16bf b, v8f c) {
  // D = A(16x32) * B(32x16) + C, fp32 accumulate
  return __builtin_amdgcn_wmma_f32_16x16x32_bf16(false, a, false, b, (short)0, c,
                                                 false, false);
}

// Async 16B global -> LDS copy (gfx1250 GLOBAL_LOAD_ASYNC_TO_LDS_B128).
// ldsOff: LDS byte offset (low 32 bits of a generic pointer to __shared__).
static __device__ __forceinline__ void async_copy_b128(unsigned ldsOff,
                                                       const void* gptr) {
  unsigned long long ga = (unsigned long long)gptr;
  asm volatile("global_load_async_to_lds_b128 %0, %1, off"
               :: "v"(ldsOff), "v"(ga) : "memory");
}
static __device__ __forceinline__ void async_wait0() {
  asm volatile("s_wait_asynccnt 0x0" ::: "memory");
}

// ---------------------------------------------------------------------------
// Kernel 1: qkv = tokens @ w_qkv, scatter into q/k/v (which, b, h, n, d) bf16
// tokens[m, k] = x[b, k, m]   (x is (b, c, n) with n = 32*32)
// ---------------------------------------------------------------------------
__global__ __launch_bounds__(128) void qkv_gemm_kernel(
    const float* __restrict__ x, const float* __restrict__ wqkv,
    __bf16* __restrict__ qkv) {
  const int colBase = blockIdx.x * 64;   // 0..2304
  const int rowBase = blockIdx.y * 64;   // 0..1024
  const int b = blockIdx.z;
  const int tid = threadIdx.x;
  const int wave = tid >> 5, lane = tid & 31;
  const int laneM = lane & 15, laneH = lane >> 4;

  __shared__ __attribute__((aligned(16))) __bf16 As[64 * 32];  // [m][k]
  __shared__ __attribute__((aligned(16))) __bf16 Bs[64 * 32];  // [j][k] (transposed)

  const float* xb = x + (size_t)b * DIMC * NTOK;

  v8f acc[4] = {{0, 0, 0, 0, 0, 0, 0, 0},
                {0, 0, 0, 0, 0, 0, 0, 0},
                {0, 0, 0, 0, 0, 0, 0, 0},
                {0, 0, 0, 0, 0, 0, 0, 0}};

  for (int kb = 0; kb < DIMC; kb += 32) {
    __syncthreads();
    // Stage A tile: As[m][k] = x[b][kb+k][rowBase+m]  (coalesced along m)
    for (int i = tid; i < 64 * 32; i += 128) {
      int k = i >> 6, m = i & 63;
      As[m * 32 + k] = (__bf16)xb[(size_t)(kb + k) * NTOK + rowBase + m];
    }
    // Stage B tile transposed: Bs[j][k] = wqkv[kb+k][colBase+j] (coalesced along j)
    for (int i = tid; i < 64 * 32; i += 128) {
      int k = i >> 6, j = i & 63;
      Bs[j * 32 + k] = (__bf16)wqkv[(size_t)(kb + k) * QKVC + colBase + j];
    }
    __syncthreads();

    AFrag a;
    const __bf16* ap = As + (wave * 16 + laneM) * 32;
    a.q[0] = *(const u32x4*)(ap + 8 * laneH);        // K = 8h + 0..7
    a.q[1] = *(const u32x4*)(ap + 16 + 8 * laneH);   // K = 16 + 8h + 0..7
#pragma unroll
    for (int t = 0; t < 4; ++t) {
      AFrag bf;
      const __bf16* bp = Bs + (t * 16 + laneM) * 32 + 8 * laneH;  // K = 8h + e, 16 + 8h + e
      bf.q[0] = *(const u32x4*)(bp);
      bf.q[1] = *(const u32x4*)(bp + 16);
      acc[t] = wmma_bf16(a.v, bf.v, acc[t]);
    }
  }

  // Scatter: column jj of 3c axis decomposes as jj = d*36 + which*12 + h
#pragma unroll
  for (int t = 0; t < 4; ++t) {
    int jj = colBase + t * 16 + laneM;
    int d = jj / 36, rem = jj % 36;
    int which = rem / 12, hh = rem % 12;
    __bf16* dst = qkv +
        (((size_t)which * NB + b) * NHEAD + hh) * (size_t)(NTOK * DHEAD) + d;
#pragma unroll
    for (int r = 0; r < 8; ++r) {
      int i = rowBase + 8 * laneH + r;
      dst[(size_t)i * DHEAD] = (__bf16)acc[t][r];
    }
  }
}

// ---------------------------------------------------------------------------
// Kernel 2: flash attention per (b, h, 64-row block); 4 waves x 16 rows
// energy = (Q K^T) * sqrt(64); online softmax; O = P V; ctx[b][i][h*64+d]
// ---------------------------------------------------------------------------
__global__ __launch_bounds__(128) void attn_kernel(
    const __bf16* __restrict__ qkv, __bf16* __restrict__ ctx) {
  const int rowBlk = blockIdx.x & 15;
  const int h = (blockIdx.x >> 4) % NHEAD;
  const int b = blockIdx.x / (16 * NHEAD);
  const int tid = threadIdx.x;
  const int wave = tid >> 5, lane = tid & 31;
  const int laneM = lane & 15, laneH = lane >> 4;

  __shared__ __attribute__((aligned(16))) __bf16 Kt[32 * 64];   // [key][d]
  __shared__ __attribute__((aligned(16))) __bf16 Vt[64 * 32];   // [d][key]
  __shared__ __attribute__((aligned(16))) __bf16 Pb[4][16 * 32];

  const size_t headMat = (size_t)NTOK * DHEAD;
  const __bf16* Q = qkv + ((size_t)0 * NB * NHEAD + (size_t)b * NHEAD + h) * headMat;
  const __bf16* K = qkv + ((size_t)1 * NB * NHEAD + (size_t)b * NHEAD + h) * headMat;
  const __bf16* V = qkv + ((size_t)2 * NB * NHEAD + (size_t)b * NHEAD + h) * headMat;

  const int rowBase = rowBlk * 64 + wave * 16;

  // Resident Q fragments: 16 rows x 64 (two K=32 chunks)
  AFrag qa[2];
  {
    const __bf16* qp = Q + (size_t)(rowBase + laneM) * DHEAD;
    qa[0].q[0] = *(const u32x4*)(qp + 8 * laneH);
    qa[0].q[1] = *(const u32x4*)(qp + 16 + 8 * laneH);
    qa[1].q[0] = *(const u32x4*)(qp + 32 + 8 * laneH);
    qa[1].q[1] = *(const u32x4*)(qp + 48 + 8 * laneH);
  }

  v8f o[4] = {{0, 0, 0, 0, 0, 0, 0, 0},
              {0, 0, 0, 0, 0, 0, 0, 0},
              {0, 0, 0, 0, 0, 0, 0, 0},
              {0, 0, 0, 0, 0, 0, 0, 0}};
  float mrow[8], lrow[8];
#pragma unroll
  for (int r = 0; r < 8; ++r) { mrow[r] = -3.0e38f; lrow[r] = 0.0f; }

  const unsigned KtOff = (unsigned)(size_t)(const void*)Kt;  // LDS byte offset

  for (int ch = 0; ch < NTOK / 32; ++ch) {
    const int keyBase = ch * 32;
    __syncthreads();
    // Stage K chunk row-major via async global->LDS (bf16 verbatim copy,
    // 256 x 16B chunks, tracked by ASYNCcnt); V chunk transposed via VALU.
    {
      const __bf16* kg = K + (size_t)keyBase * DHEAD;
      for (int i = tid; i < 256; i += 128)
        async_copy_b128(KtOff + (unsigned)i * 16u, kg + i * 8);
      const __bf16* vg = V + (size_t)keyBase * DHEAD;
      for (int i = tid; i < 32 * 64; i += 128) {
        int key = i >> 6, d = i & 63;
        Vt[d * 32 + key] = vg[i];
      }
      async_wait0();
    }
    __syncthreads();

    // S tiles (16 rows x 32 keys): 2 column tiles, K-dim = d = 64 (2 WMMA each)
    v8f s[2] = {{0, 0, 0, 0, 0, 0, 0, 0}, {0, 0, 0, 0, 0, 0, 0, 0}};
#pragma unroll
    for (int t = 0; t < 2; ++t) {
      const __bf16* kp = Kt + (t * 16 + laneM) * 64 + 8 * laneH;
      AFrag b0, b1;
      b0.q[0] = *(const u32x4*)(kp);       b0.q[1] = *(const u32x4*)(kp + 16);
      b1.q[0] = *(const u32x4*)(kp + 32);  b1.q[1] = *(const u32x4*)(kp + 48);
      s[t] = wmma_bf16(qa[0].v, b0.v, s[t]);
      s[t] = wmma_bf16(qa[1].v, b1.v, s[t]);
    }

    // Online softmax: row M = r + 8*laneH lives at element r, columns across
    // the 16-lane group -> xor-reduce with masks 1,2,4,8 stays in-group.
#pragma unroll
    for (int r = 0; r < 8; ++r) {
      float s0 = s[0][r] * 8.0f;  // scale = sqrt(d_head)
      float s1 = s[1][r] * 8.0f;
      float mx = fmaxf(s0, s1);
      mx = fmaxf(mx, __shfl_xor(mx, 1, 32));
      mx = fmaxf(mx, __shfl_xor(mx, 2, 32));
      mx = fmaxf(mx, __shfl_xor(mx, 4, 32));
      mx = fmaxf(mx, __shfl_xor(mx, 8, 32));
      float mn = fmaxf(mrow[r], mx);
      float alpha = __expf(mrow[r] - mn);
      float p0 = __expf(s0 - mn);
      float p1 = __expf(s1 - mn);
      float rs = p0 + p1;
      rs += __shfl_xor(rs, 1, 32);
      rs += __shfl_xor(rs, 2, 32);
      rs += __shfl_xor(rs, 4, 32);
      rs += __shfl_xor(rs, 8, 32);
      lrow[r] = lrow[r] * alpha + rs;
      mrow[r] = mn;
      s[0][r] = p0; s[1][r] = p1;
      o[0][r] *= alpha; o[1][r] *= alpha; o[2][r] *= alpha; o[3][r] *= alpha;
    }

    // P: C-layout -> A-fragment layout via per-wave LDS tile (16x32 bf16)
    __bf16* pw = Pb[wave];
#pragma unroll
    for (int t = 0; t < 2; ++t)
#pragma unroll
      for (int r = 0; r < 8; ++r)
        pw[(r + 8 * laneH) * 32 + t * 16 + laneM] = (__bf16)s[t][r];
    asm volatile("" ::: "memory");  // keep DS store->load order (HW DS in-order per wave)
    AFrag pa;
    {
      const __bf16* pr = pw + laneM * 32;
      pa.q[0] = *(const u32x4*)(pr + 8 * laneH);
      pa.q[1] = *(const u32x4*)(pr + 16 + 8 * laneH);
    }

    // O(16x64) += P(16x32) * V(32x64): 4 d-tiles, K=32 keys
#pragma unroll
    for (int dt = 0; dt < 4; ++dt) {
      AFrag bv;
      const __bf16* vp = Vt + (dt * 16 + laneM) * 32 + 8 * laneH;
      bv.q[0] = *(const u32x4*)(vp);
      bv.q[1] = *(const u32x4*)(vp + 16);
      o[dt] = wmma_bf16(pa.v, bv.v, o[dt]);
    }
  }

  // Normalize and write ctx[b][i][h*64 + d] (bf16)
#pragma unroll
  for (int r = 0; r < 8; ++r) lrow[r] = 1.0f / lrow[r];
#pragma unroll
  for (int dt = 0; dt < 4; ++dt)
#pragma unroll
    for (int r = 0; r < 8; ++r) {
      int i = rowBase + r + 8 * laneH;
      ctx[((size_t)b * NTOK + i) * DIMC + h * DHEAD + dt * 16 + laneM] =
          (__bf16)(o[dt][r] * lrow[r]);
    }
}

// ---------------------------------------------------------------------------
// Kernel 3: out = ctx @ w_proj, written transposed as (b, c, n) fp32
// ---------------------------------------------------------------------------
__global__ __launch_bounds__(128) void proj_gemm_kernel(
    const __bf16* __restrict__ ctx, const float* __restrict__ wproj,
    float* __restrict__ out) {
  const int colBase = blockIdx.x * 64;   // 0..768
  const int rowBase = blockIdx.y * 64;   // 0..1024
  const int b = blockIdx.z;
  const int tid = threadIdx.x;
  const int wave = tid >> 5, lane = tid & 31;
  const int laneM = lane & 15, laneH = lane >> 4;

  __shared__ __attribute__((aligned(16))) __bf16 As[64 * 32];  // [m][k]
  __shared__ __attribute__((aligned(16))) __bf16 Bs[64 * 32];  // [j][k]

  const __bf16* ab = ctx + (size_t)b * NTOK * DIMC;
  const unsigned AsOff = (unsigned)(size_t)(const void*)As;  // LDS byte offset

  v8f acc[4] = {{0, 0, 0, 0, 0, 0, 0, 0},
                {0, 0, 0, 0, 0, 0, 0, 0},
                {0, 0, 0, 0, 0, 0, 0, 0},
                {0, 0, 0, 0, 0, 0, 0, 0}};

  for (int kb = 0; kb < DIMC; kb += 32) {
    __syncthreads();
    // Stage A (already bf16): async global->LDS 16B copies
    for (int i = tid; i < 256; i += 128) {
      int m = i >> 2, koff = (i & 3) * 8;
      async_copy_b128(AsOff + (unsigned)i * 16u,
                      ab + (size_t)(rowBase + m) * DIMC + kb + koff);
    }
    // Stage B transposed: Bs[j][k] = wproj[kb+k][colBase+j]
    for (int i = tid; i < 64 * 32; i += 128) {
      int k = i >> 6, j = i & 63;
      Bs[j * 32 + k] = (__bf16)wproj[(size_t)(kb + k) * DIMC + colBase + j];
    }
    async_wait0();
    __syncthreads();

    AFrag a;
    const __bf16* ap = As + (wave * 16 + laneM) * 32;
    a.q[0] = *(const u32x4*)(ap + 8 * laneH);
    a.q[1] = *(const u32x4*)(ap + 16 + 8 * laneH);
#pragma unroll
    for (int t = 0; t < 4; ++t) {
      AFrag bf;
      const __bf16* bp = Bs + (t * 16 + laneM) * 32 + 8 * laneH;
      bf.q[0] = *(const u32x4*)(bp);
      bf.q[1] = *(const u32x4*)(bp + 16);
      acc[t] = wmma_bf16(a.v, bf.v, acc[t]);
    }
  }

  // out[b][cout][i]: per lane cout fixed, 8 consecutive rows i -> two float4s
#pragma unroll
  for (int t = 0; t < 4; ++t) {
    int cout = colBase + t * 16 + laneM;
    float* dst = out + ((size_t)b * DIMC + cout) * NTOK + rowBase + 8 * laneH;
    Acc8 u; u.v = acc[t];
    *(f32x4*)(dst) = u.f[0];
    *(f32x4*)(dst + 4) = u.f[1];
  }
}

// ---------------------------------------------------------------------------
extern "C" void kernel_launch(void* const* d_in, const int* in_sizes, int n_in,
                              void* d_out, int out_size, void* d_ws, size_t ws_size,
                              hipStream_t stream) {
  const float* x     = (const float*)d_in[0];  // (8, 768, 32, 32)
  const float* wqkv  = (const float*)d_in[1];  // (768, 2304)
  const float* wproj = (const float*)d_in[2];  // (768, 768)
  float* out = (float*)d_out;                  // (8, 768, 32, 32)

  // Workspace: qkv bf16 (3*8*12*1024*64) then ctx bf16 (8*1024*768)
  const size_t qkvBytes = (size_t)3 * NB * NHEAD * NTOK * DHEAD * 2;
  __bf16* qkv = (__bf16*)d_ws;
  __bf16* ctx = (__bf16*)((char*)d_ws + qkvBytes);

  dim3 blk(128);
  qkv_gemm_kernel<<<dim3(QKVC / 64, NTOK / 64, NB), blk, 0, stream>>>(x, wqkv, qkv);
  attn_kernel<<<dim3(NB * NHEAD * (NTOK / 64)), blk, 0, stream>>>(qkv, ctx);
  proj_gemm_kernel<<<dim3(DIMC / 64, NTOK / 64, NB), blk, 0, stream>>>(ctx, wproj, out);
}
